// TorchLSTM_833223655601
// MI455X (gfx1250) — compile-verified, matrix operand order checked by reading, not run on hardware
//
#include <hip/hip_runtime.h>
#include <hip/hip_bf16.h>

#define BB 4096
#define TT 256
#define FF 32

typedef __bf16 bf16;
typedef __attribute__((ext_vector_type(16))) __bf16 v16bf;
typedef __attribute__((ext_vector_type(8)))  __bf16 v8bf;
typedef __attribute__((ext_vector_type(8)))  float  v8f;
typedef __attribute__((ext_vector_type(4)))  float  fl4;

static __device__ __forceinline__ v16bf cat16(v8bf lo, v8bf hi) {
  return __builtin_shufflevector(lo, hi, 0,1,2,3,4,5,6,7,8,9,10,11,12,13,14,15);
}

static __device__ __forceinline__ v8f wmma_bf16(v16bf a, v16bf b, v8f c) {
  return __builtin_amdgcn_wmma_f32_16x16x32_bf16(false, a, false, b, (short)0, c,
                                                 false, false);
}

// A-fragment (16x32 bf16 chunk) from row-major LDS scratch.
// lanes 0-15: row r, K runs {k0..k0+7, k0+16..k0+23}; lanes 16-31: +8.
static __device__ __forceinline__ v16bf load_a2(const bf16* rowbase, int k0) {
  v8bf lo = *(const v8bf*)(rowbase + k0);
  v8bf hi = *(const v8bf*)(rowbase + k0 + 16);
  return cat16(lo, hi);
}

// Build one 32-K B-fragment for this lane straight from a global fp32 weight
// row (W is [N,K] row-major; column n is fixed per lane). Element e of the
// fragment maps to k = kbase + hiHalf*8 + (e<8 ? e : e+8): two contiguous
// 8-float runs -> four fl4 loads + bf16 convert. Done ONCE, lives in VGPRs.
static __device__ __forceinline__ v16bf load_wfrag(const float* __restrict__ Wrow,
                                                   int kbase, int hiHalf) {
  const float* p = Wrow + kbase + hiHalf * 8;
  fl4 a0 = *(const fl4*)(p);
  fl4 a1 = *(const fl4*)(p + 4);
  fl4 a2 = *(const fl4*)(p + 16);
  fl4 a3 = *(const fl4*)(p + 20);
  v16bf b;
  #pragma unroll
  for (int i = 0; i < 4; ++i) {
    b[i]      = (bf16)a0[i];
    b[4 + i]  = (bf16)a1[i];
    b[8 + i]  = (bf16)a2[i];
    b[12 + i] = (bf16)a3[i];
  }
  return b;
}

static __device__ __forceinline__ v8f splat8(float b) {
  v8f v;
  #pragma unroll
  for (int i = 0; i < 8; ++i) v[i] = b;
  return v;
}

static __device__ __forceinline__ v16bf zero16() {
  v16bf v;
  #pragma unroll
  for (int i = 0; i < 16; ++i) v[i] = (bf16)0.0f;
  return v;
}

// CDNA5 has hardware V_TANH_F32 (TRANS32 pipe): 1 trans op per tanh, and
// sigmoid(x) = 0.5*tanh(x/2) + 0.5 -> 1 trans + mul + fma. This halves the
// pressure on the transcendental pipe, which is the bottleneck here.
static __device__ __forceinline__ float fast_tanh(float x) {
  return __builtin_amdgcn_tanhf(x);
}
static __device__ __forceinline__ float fast_sigmoid(float x) {
  return __builtin_fmaf(0.5f, __builtin_amdgcn_tanhf(0.5f * x), 0.5f);
}

__global__ __launch_bounds__(128, 1)
void lstm2_fused_wmma(const float* __restrict__ x,
                      const float* __restrict__ W_ih1, const float* __restrict__ W_hh1,
                      const float* __restrict__ b_ih1, const float* __restrict__ b_hh1,
                      const float* __restrict__ W_ih2, const float* __restrict__ W_hh2,
                      const float* __restrict__ b_ih2, const float* __restrict__ b_hh2,
                      const float* __restrict__ W_head, const float* __restrict__ b_head,
                      float* __restrict__ out)
{
  extern __shared__ char smem_raw[];
  bf16* sH1 = (bf16*)smem_raw;          // 2 tiles * 16*64 bf16 (4 KB)
  bf16* sH2 = sH1 + 2 * 16 * 64;        // 2 tiles * 16*32 bf16 (2 KB)

  const int tid    = threadIdx.x;
  const int lane   = tid & 31;
  const int wv     = tid >> 5;       // wave 0..3
  const int sub    = wv & 1;         // N-split partner within an M-tile
  const int mt     = wv >> 1;        // which M-tile (16 batch rows) this wave serves
  const int lnLow  = lane & 15;
  const int hiHalf = lane >> 4;      // 0|1
  const int hiRow  = hiHalf * 8;     // C/D row offset for high lanes
  const int k0x    = hiHalf * 8;     // A-layout K base for this half-wave

  // ---- hoist ALL loop-invariant B-fragments into VGPRs (once) ----
  // Layer 1: Wc1[k][n], k<32 -> W_ih1[n][k], else W_hh1[n][k-32]. N=256, K=96.
  // This wave owns hidden groups {2*sub, 2*sub+1}: 2x4x3 = 24 fragments.
  v16bf w1[2][4][3];
  #pragma unroll
  for (int gl = 0; gl < 2; ++gl) {
    const int gG = sub * 2 + gl;
    #pragma unroll
    for (int q = 0; q < 4; ++q) {
      const int n = q * 64 + gG * 16 + lnLow;     // B-column for this lane
      w1[gl][q][0] = load_wfrag(W_ih1 + n * 32, 0,  hiHalf);
      w1[gl][q][1] = load_wfrag(W_hh1 + n * 64, 0,  hiHalf);
      w1[gl][q][2] = load_wfrag(W_hh1 + n * 64, 32, hiHalf);
    }
  }
  // Layer 2: Wc2[k][n], k<64 -> W_ih2[n][k], else W_hh2[n][k-64]. N=128, K=96.
  // This wave owns hidden group sub: 4x3 = 12 fragments.
  v16bf w2[4][3];
  #pragma unroll
  for (int q = 0; q < 4; ++q) {
    const int n = q * 32 + sub * 16 + lnLow;
    w2[q][0] = load_wfrag(W_ih2 + n * 64, 0,  hiHalf);
    w2[q][1] = load_wfrag(W_ih2 + n * 64, 32, hiHalf);
    w2[q][2] = load_wfrag(W_hh2 + n * 32, 0,  hiHalf);
  }

  // ---- hoist biases: every accumulator element of a lane shares column N ----
  float bias1[2][4];
  #pragma unroll
  for (int gl = 0; gl < 2; ++gl) {
    const int gG = sub * 2 + gl;
    #pragma unroll
    for (int q = 0; q < 4; ++q) {
      const int n = q * 64 + gG * 16 + lnLow;
      bias1[gl][q] = b_ih1[n] + b_hh1[n];
    }
  }
  float bias2[4];
  #pragma unroll
  for (int q = 0; q < 4; ++q) {
    const int n = q * 32 + sub * 16 + lnLow;
    bias2[q] = b_ih2[n] + b_hh2[n];
  }

  const int rowg = blockIdx.x * 32 + mt * 16 + lnLow;
  const float* xrow = x + (size_t)rowg * TT * FF;

  bf16* myH1 = sH1 + mt * (16 * 64);
  bf16* myH2 = sH2 + mt * (16 * 32);

  v16bf aH10 = zero16(), aH11 = zero16(), aH2 = zero16();
  v8f c1v[2]; c1v[0] = splat8(0.0f); c1v[1] = splat8(0.0f);
  v8f c2v = splat8(0.0f);

  for (int t = 0; t < TT; ++t) {
    // ---- x_t A-fragment (global fp32 -> bf16, A-layout) ----
    const float* xr = xrow + t * FF;
    fl4 x0 = *(const fl4*)(xr + k0x);
    fl4 x1 = *(const fl4*)(xr + k0x + 4);
    fl4 x2 = *(const fl4*)(xr + k0x + 16);
    fl4 x3 = *(const fl4*)(xr + k0x + 20);
    if (t + 1 < TT) __builtin_prefetch(xr + FF, 0, 1);
    v16bf aX;
    #pragma unroll
    for (int i = 0; i < 4; ++i) {
      aX[i]      = (bf16)x0[i];
      aX[4 + i]  = (bf16)x1[i];
      aX[8 + i]  = (bf16)x2[i];
      aX[12 + i] = (bf16)x3[i];
    }

    // ---- layer 1: two gate-groups per wave; 24 back-to-back WMMAs ----
    #pragma unroll
    for (int gl = 0; gl < 2; ++gl) {
      const int gG = sub * 2 + gl;
      v8f aI = splat8(bias1[gl][0]);
      v8f aF = splat8(bias1[gl][1]);
      v8f aG = splat8(bias1[gl][2]);
      v8f aO = splat8(bias1[gl][3]);
      #pragma unroll
      for (int kc = 0; kc < 3; ++kc) {
        v16bf a = (kc == 0) ? aX : ((kc == 1) ? aH10 : aH11);
        aI = wmma_bf16(a, w1[gl][0][kc], aI);
        aF = wmma_bf16(a, w1[gl][1][kc], aF);
        aG = wmma_bf16(a, w1[gl][2][kc], aG);
        aO = wmma_bf16(a, w1[gl][3][kc], aO);
      }
      bf16* hdst = myH1 + gG * 16 + lnLow;
      #pragma unroll
      for (int j = 0; j < 8; ++j) {
        float iv = fast_sigmoid(aI[j]);
        float fv = fast_sigmoid(aF[j]);
        float gv = fast_tanh(aG[j]);
        float ov = fast_sigmoid(aO[j]);
        float cn = fv * c1v[gl][j] + iv * gv;
        c1v[gl][j] = cn;
        hdst[(j + hiRow) * 64] = (bf16)(ov * fast_tanh(cn));
      }
    }
    __syncthreads();   // h1_t visible; prior readers of sH1 done

    // h1_t A-fragments: feed layer-2 now and layer-1 recurrence next step
    aH10 = load_a2(myH1 + lnLow * 64, k0x);
    aH11 = load_a2(myH1 + lnLow * 64, 32 + k0x);

    // ---- layer 2: one gate-group per wave; 12 back-to-back WMMAs ----
    {
      const int g2 = sub;
      v8f aI = splat8(bias2[0]);
      v8f aF = splat8(bias2[1]);
      v8f aG = splat8(bias2[2]);
      v8f aO = splat8(bias2[3]);
      #pragma unroll
      for (int kc = 0; kc < 3; ++kc) {
        v16bf a = (kc == 0) ? aH10 : ((kc == 1) ? aH11 : aH2);
        aI = wmma_bf16(a, w2[0][kc], aI);
        aF = wmma_bf16(a, w2[1][kc], aF);
        aG = wmma_bf16(a, w2[2][kc], aG);
        aO = wmma_bf16(a, w2[3][kc], aO);
      }
      bf16* hdst = myH2 + g2 * 16 + lnLow;
      #pragma unroll
      for (int j = 0; j < 8; ++j) {
        float iv = fast_sigmoid(aI[j]);
        float fv = fast_sigmoid(aF[j]);
        float gv = fast_tanh(aG[j]);
        float ov = fast_sigmoid(aO[j]);
        float cn = fv * c2v[j] + iv * gv;
        c2v[j] = cn;
        hdst[(j + hiRow) * 32] = (bf16)(ov * fast_tanh(cn));
      }
    }
    __syncthreads();   // h2_t visible; separates aH1 reads from next-step stores
    aH2 = load_a2(myH2 + lnLow * 32, k0x);
  }

  // ---- head: out = h2_last @ W_head.T + b_head  (tiny, scalar) ----
  if (sub == 0 && lane < 16) {
    const int m = lane;
    const bf16* hrow = myH2 + m * 32;
    const int orow = blockIdx.x * 32 + mt * 16 + m;
    #pragma unroll
    for (int o = 0; o < 3; ++o) {
      float s = b_head[o];
      #pragma unroll
      for (int k = 0; k < 32; ++k) s += (float)hrow[k] * W_head[o * 32 + k];
      out[orow * 3 + o] = s;
    }
  }
}

extern "C" void kernel_launch(void* const* d_in, const int* in_sizes, int n_in,
                              void* d_out, int out_size, void* d_ws, size_t ws_size,
                              hipStream_t stream) {
  const float* x      = (const float*)d_in[0];
  const float* W_ih1  = (const float*)d_in[1];
  const float* W_hh1  = (const float*)d_in[2];
  const float* b_ih1  = (const float*)d_in[3];
  const float* b_hh1  = (const float*)d_in[4];
  const float* W_ih2  = (const float*)d_in[5];
  const float* W_hh2  = (const float*)d_in[6];
  const float* b_ih2  = (const float*)d_in[7];
  const float* b_hh2  = (const float*)d_in[8];
  const float* W_head = (const float*)d_in[9];
  const float* b_head = (const float*)d_in[10];
  float* out = (float*)d_out;

  const size_t shmem = (2 * 16 * 64 + 2 * 16 * 32) * sizeof(bf16); // 6144 B

  dim3 grid(BB / 32);   // 128 workgroups, each owns 32 batch rows
  dim3 block(128);      // 4 wave32
  lstm2_fused_wmma<<<grid, block, shmem, stream>>>(
      x, W_ih1, W_hh1, b_ih1, b_hh1, W_ih2, W_hh2, b_ih2, b_hh2,
      W_head, b_head, out);
}
